// KSpaceMambaBlock_31593779429659
// MI455X (gfx1250) — compile-verified, matrix operand order checked by reading, not run on hardware
//
#include <hip/hip_runtime.h>

typedef __bf16 bf16;
typedef __bf16 v16bf __attribute__((ext_vector_type(16)));
typedef float  v8f   __attribute__((ext_vector_type(8)));

#define LSEQ 8192
#define DM   128
#define ED   256
#define NST  16
#define NB_  8
#define NCH  64     // scan chunks
#define CLCH 128    // chunk length

static __device__ __forceinline__ float sigm(float x) { return 1.f / (1.f + __expf(-x)); }
static __device__ __forceinline__ float silu(float x) { return x * sigm(x); }

// ---------------------------------------------------------------- scalars per position
__global__ void k_scalars(const float* __restrict__ mask,
                          const float* __restrict__ edt, const float* __restrict__ eB,
                          const float* __restrict__ eC,  const float* __restrict__ wr,
                          const float* __restrict__ wm,
                          float* __restrict__ m_arr, float* __restrict__ sdt,
                          float* __restrict__ sB, float* __restrict__ sC) {
  int l = blockIdx.x * 256 + threadIdx.x;
  if (l >= LSEQ) return;
  int gx = l >> 8, gy = (l >> 3) & 31, gz = l & 7;
  float s = 0.f;
#pragma unroll
  for (int i = 0; i < 2; ++i)
#pragma unroll
    for (int j = 0; j < 2; ++j)
#pragma unroll
      for (int k = 0; k < 2; ++k)
        s += mask[((2 * gx + i) * 64 + (2 * gy + j)) * 16 + (2 * gz + k)];
  float m = s * 0.125f;
  m = fminf(fmaxf(m, 0.f), 1.f);
  float rho = (float)l * (1.f / (float)(LSEQ - 1));
  int band = (int)floorf(rho * (float)NB_);
  band = band < 0 ? 0 : (band > NB_ - 1 ? NB_ - 1 : band);
  m_arr[l] = m;
  sdt[l] = 2.f * sigm(edt[band] + wr[0] * rho + wm[0] * m);
  sB[l]  = 2.f * sigm(eB[band]  + wr[1] * rho + wm[1] * m);
  sC[l]  = 2.f * sigm(eC[band]  + wr[2] * rho + wm[2] * m);
}

// ---------------------------------------------------------------- gather patches -> bf16 [L][128]
__global__ void k_gather(const float* __restrict__ ks, bf16* __restrict__ Xin) {
  int idx = blockIdx.x * 256 + threadIdx.x;
  if (idx >= LSEQ * 128) return;
  int q = idx & 127, l = idx >> 7;
  int gx = l >> 8, gy = (l >> 3) & 31, gz = l & 7;
  int c = q >> 3, i = (q >> 2) & 1, j = (q >> 1) & 1, k = q & 1;
  Xin[idx] = (bf16)ks[((c * 64 + (2 * gx + i)) * 64 + (2 * gy + j)) * 16 + (2 * gz + k)];
}

// ---------------------------------------------------------------- weight casts
__global__ void k_cast_direct(const float* __restrict__ in, bf16* __restrict__ out, int n) {
  int i = blockIdx.x * 256 + threadIdx.x;
  if (i < n) out[i] = (bf16)in[i];
}
// in f32 [K][Nsrc] -> out bf16 [Ndst][K] (transposed, zero-padded rows)
__global__ void k_castT(const float* __restrict__ in, bf16* __restrict__ out,
                        int K, int Nsrc, int Ndst) {
  int i = blockIdx.x * 256 + threadIdx.x;
  if (i >= Ndst * K) return;
  int n = i / K, k = i - n * K;
  out[i] = (n < Nsrc) ? (bf16)in[(size_t)k * Nsrc + n] : (bf16)0.f;
}

// ---------------------------------------------------------------- generic WMMA GEMM
// One wave computes a 16 x (JT*16) strip: the A-fragment is loaded once per K-step
// and reused across JT WMMAs. JT and MODE are compile-time so the inner loop is
// straight-line (uniform EXEC for WMMA, pipelined b128 loads).
// A bf16 [M][K], Bt bf16 [N][K].
// MODE: 0 plain(+bias), 1 split+silu, 2 residual+bf16, 3 kspace scatter
template <int JT, int MODE>
__global__ __launch_bounds__(256) void gemm_wmma(
    const bf16* __restrict__ A, const bf16* __restrict__ Bt, float* __restrict__ C,
    int M, int N, int K,
    const float* __restrict__ bias, const float* __restrict__ res,
    float* __restrict__ out1, bf16* __restrict__ outb, const float* __restrict__ bias2) {
  const int lane = threadIdx.x & 31;
  const int wave = threadIdx.x >> 5;
  const int groupsN = (N + 63) >> 6;
  const int tile = blockIdx.x * 8 + wave;
  if (tile >= (M >> 4) * groupsN) return;
  const int m0 = (tile / groupsN) << 4;
  const int nbase = (tile % groupsN) << 6;
  const int sub = lane & 15, half = lane >> 4;
  const bf16* ap = A + (size_t)(m0 + sub) * K + half * 8;
  const bf16* bp[JT];
#pragma unroll
  for (int j = 0; j < JT; ++j)
    bp[j] = Bt + (size_t)(nbase + j * 16 + sub) * K + half * 8;
  v8f acc[JT];
#pragma unroll
  for (int j = 0; j < JT; ++j) acc[j] = (v8f){0.f, 0.f, 0.f, 0.f, 0.f, 0.f, 0.f, 0.f};
  for (int k0 = 0; k0 < K; k0 += 32) {
    union { uint4 u[2]; v16bf v; } fa;
    fa.u[0] = *(const uint4*)(ap + k0);
    fa.u[1] = *(const uint4*)(ap + k0 + 16);
    __builtin_prefetch((const void*)(ap + k0 + 32), 0, 1);
#pragma unroll
    for (int j = 0; j < JT; ++j) {
      union { uint4 u[2]; v16bf v; } fb;
      fb.u[0] = *(const uint4*)(bp[j] + k0);
      fb.u[1] = *(const uint4*)(bp[j] + k0 + 16);
      __builtin_prefetch((const void*)(bp[j] + k0 + 32), 0, 1);
      acc[j] = __builtin_amdgcn_wmma_f32_16x16x32_bf16(false, fa.v, false, fb.v,
                                                       (short)0, acc[j], false, false);
    }
  }
#pragma unroll
  for (int j = 0; j < JT; ++j) {
    int n = nbase + j * 16 + sub;
#pragma unroll
    for (int r = 0; r < 8; ++r) {
      int m = m0 + r + half * 8;
      float v = acc[j][r];
      if constexpr (MODE == 0) {
        if (bias) v += bias[n];
        C[(size_t)m * N + n] = v;
      } else if constexpr (MODE == 1) {          // xz split: xp | silu(z)
        v += bias[n];
        if (n < ED) C[(size_t)m * ED + n] = v;
        else        out1[(size_t)m * ED + (n - ED)] = silu(v);
      } else if constexpr (MODE == 2) {          // h2 = h + y@Wout + bout
        float t = res[(size_t)m * DM + n] + v + bias[n];
        C[(size_t)m * DM + n] = t;
        outb[(size_t)m * DM + n] = (bf16)t;
      } else {                                   // unembed scatter into k-space
        int gx = m >> 8, gy = (m >> 3) & 31, gz = m & 7;
        int c = n >> 3, i = (n >> 2) & 1, jj = (n >> 1) & 1, kk = n & 1;
        size_t idx = (((size_t)c * 64 + (2 * gx + i)) * 64 + (2 * gy + jj)) * 16 + (2 * gz + kk);
        C[idx] = res[idx] + v + bias2[c];
      }
    }
  }
}

// ---------------------------------------------------------------- double LayerNorm (row = 128), wave per row
__global__ __launch_bounds__(256) void k_ln(
    const float* __restrict__ xraw, const float* __restrict__ pe_b,
    const float* __restrict__ cond_band, const float* __restrict__ cond_rad_w,
    const float* __restrict__ cond_mask_w, const float* __restrict__ m_arr,
    const float* __restrict__ ln_g, const float* __restrict__ ln_b,
    const float* __restrict__ fn_g, const float* __restrict__ fn_b,
    float* __restrict__ h, bf16* __restrict__ hln) {
  int l = blockIdx.x * 8 + (threadIdx.x >> 5);
  if (l >= LSEQ) return;
  int lane = threadIdx.x & 31;
  int d0 = lane * 4;
  float rho = (float)l * (1.f / (float)(LSEQ - 1));
  int band = (int)floorf(rho * (float)NB_);
  band = band < 0 ? 0 : (band > NB_ - 1 ? NB_ - 1 : band);
  float mv = m_arr[l];
  float v[4];
#pragma unroll
  for (int i = 0; i < 4; ++i) {
    int d = d0 + i;
    v[i] = xraw[(size_t)l * DM + d] + pe_b[d] +
           cond_band[band * DM + d] + rho * cond_rad_w[d] + mv * cond_mask_w[d];
  }
  float s = v[0] + v[1] + v[2] + v[3];
#pragma unroll
  for (int o = 16; o >= 1; o >>= 1) s += __shfl_xor(s, o, 32);
  float mu = s * (1.f / DM);
  float q = 0.f;
#pragma unroll
  for (int i = 0; i < 4; ++i) { float d = v[i] - mu; q += d * d; }
#pragma unroll
  for (int o = 16; o >= 1; o >>= 1) q += __shfl_xor(q, o, 32);
  float inv = rsqrtf(q * (1.f / DM) + 1e-5f);
  float hv[4];
#pragma unroll
  for (int i = 0; i < 4; ++i) {
    int d = d0 + i;
    hv[i] = (v[i] - mu) * inv * ln_g[d] + ln_b[d];
    h[(size_t)l * DM + d] = hv[i];
  }
  // second LN with fn params
  float s2 = hv[0] + hv[1] + hv[2] + hv[3];
#pragma unroll
  for (int o = 16; o >= 1; o >>= 1) s2 += __shfl_xor(s2, o, 32);
  float mu2 = s2 * (1.f / DM);
  float q2 = 0.f;
#pragma unroll
  for (int i = 0; i < 4; ++i) { float d = hv[i] - mu2; q2 += d * d; }
#pragma unroll
  for (int o = 16; o >= 1; o >>= 1) q2 += __shfl_xor(q2, o, 32);
  float inv2 = rsqrtf(q2 * (1.f / DM) + 1e-5f);
#pragma unroll
  for (int i = 0; i < 4; ++i) {
    int d = d0 + i;
    hln[(size_t)l * DM + d] = (bf16)((hv[i] - mu2) * inv2 * fn_g[d] + fn_b[d]);
  }
}

// ---------------------------------------------------------------- causal depthwise conv + silu
__global__ void k_conv(const float* __restrict__ xp, const float* __restrict__ convw,
                       const float* __restrict__ convb, float* __restrict__ xc,
                       bf16* __restrict__ xcbf) {
  int idx = blockIdx.x * 256 + threadIdx.x;
  if (idx >= LSEQ * ED) return;
  int e = idx & (ED - 1), l = idx >> 8;
  float acc = convb[e];
#pragma unroll
  for (int k = 0; k < 4; ++k) {
    int ls = l + k - 3;
    if (ls >= 0) acc += xp[(size_t)ls * ED + e] * convw[e * 4 + k];
  }
  float sv = silu(acc);
  xc[idx] = sv;
  xcbf[idx] = (bf16)sv;
}

// ---------------------------------------------------------------- dt / Bm / Cm
__global__ void k_dtbc(const float* __restrict__ dbl, const float* __restrict__ Wdt,
                       const float* __restrict__ bdt, const float* __restrict__ sdt,
                       const float* __restrict__ sB, const float* __restrict__ sC,
                       float* __restrict__ dt, float* __restrict__ Bm, float* __restrict__ Cm) {
  int idx = blockIdx.x * 256 + threadIdx.x;
  if (idx >= LSEQ * ED) return;
  int e = idx & (ED - 1), l = idx >> 8;
  float a = bdt[e];
#pragma unroll
  for (int r = 0; r < 8; ++r) a += dbl[(size_t)l * 48 + r] * Wdt[r * ED + e];
  float sp = (a > 20.f) ? a : __logf(1.f + __expf(a));
  dt[idx] = sp * sdt[l];
  if (e < NST)            Bm[l * NST + e]          = dbl[(size_t)l * 48 + 8 + e]          * sB[l];
  else if (e < 2 * NST)   Cm[l * NST + (e - NST)]  = dbl[(size_t)l * 48 + 24 + (e - NST)] * sC[l];
}

// ---------------------------------------------------------------- scan pass 1: per-chunk P, S
__global__ __launch_bounds__(256) void k_scan1(
    const float* __restrict__ dt, const float* __restrict__ xc,
    const float* __restrict__ Bm, const float* __restrict__ Alog,
    float* __restrict__ Pc, float* __restrict__ Sc) {
  __shared__ float sBm[CLCH * NST];
  int c = blockIdx.x, dir = blockIdx.y, e = threadIdx.x;
  for (int idx = threadIdx.x; idx < CLCH * NST; idx += 256) {
    int s = idx >> 4, n = idx & 15;
    int l = dir ? (LSEQ - 1 - (c * CLCH + s)) : (c * CLCH + s);
    sBm[idx] = Bm[l * NST + n];
  }
  __syncthreads();
  float A[NST], P[NST], S[NST];
#pragma unroll
  for (int n = 0; n < NST; ++n) { A[n] = -__expf(Alog[e * NST + n]); P[n] = 1.f; S[n] = 0.f; }
  for (int s = 0; s < CLCH; ++s) {
    int l = dir ? (LSEQ - 1 - (c * CLCH + s)) : (c * CLCH + s);
    float dte = dt[(size_t)l * ED + e];
    float xce = xc[(size_t)l * ED + e];
#pragma unroll
    for (int n = 0; n < NST; ++n) {
      float dA = __expf(dte * A[n]);
      float dBu = (dA - 1.f) / A[n] * sBm[s * NST + n] * xce;
      S[n] = dA * S[n] + dBu;
      P[n] *= dA;
    }
  }
  size_t base = (((size_t)dir * NCH + c) * ED + e) * NST;
#pragma unroll
  for (int n = 0; n < NST; ++n) { Pc[base + n] = P[n]; Sc[base + n] = S[n]; }
}

// ---------------------------------------------------------------- scan pass 2: chunk prefix combine
__global__ __launch_bounds__(256) void k_scan2(const float* __restrict__ Pc,
                                               const float* __restrict__ Sc,
                                               float* __restrict__ Hc) {
  int dir = blockIdx.y, e = threadIdx.x;
  float H[NST];
#pragma unroll
  for (int n = 0; n < NST; ++n) H[n] = 0.f;
  for (int c = 0; c < NCH; ++c) {
    size_t base = (((size_t)dir * NCH + c) * ED + e) * NST;
#pragma unroll
    for (int n = 0; n < NST; ++n) {
      Hc[base + n] = H[n];
      H[n] = Pc[base + n] * H[n] + Sc[base + n];
    }
  }
}

// ---------------------------------------------------------------- scan pass 3: replay + emit y
__global__ __launch_bounds__(256) void k_scan3(
    const float* __restrict__ dt, const float* __restrict__ xc,
    const float* __restrict__ Bm, const float* __restrict__ Cm,
    const float* __restrict__ Alog, const float* __restrict__ Hc,
    float* __restrict__ yf, float* __restrict__ yb) {
  __shared__ float sBm[CLCH * NST];
  __shared__ float sCm[CLCH * NST];
  int c = blockIdx.x, dir = blockIdx.y, e = threadIdx.x;
  for (int idx = threadIdx.x; idx < CLCH * NST; idx += 256) {
    int s = idx >> 4, n = idx & 15;
    int l = dir ? (LSEQ - 1 - (c * CLCH + s)) : (c * CLCH + s);
    sBm[idx] = Bm[l * NST + n];
    sCm[idx] = Cm[l * NST + n];
  }
  __syncthreads();
  float A[NST], H[NST];
  size_t base = (((size_t)dir * NCH + c) * ED + e) * NST;
#pragma unroll
  for (int n = 0; n < NST; ++n) { A[n] = -__expf(Alog[e * NST + n]); H[n] = Hc[base + n]; }
  float* yout = dir ? yb : yf;
  for (int s = 0; s < CLCH; ++s) {
    int l = dir ? (LSEQ - 1 - (c * CLCH + s)) : (c * CLCH + s);
    float dte = dt[(size_t)l * ED + e];
    float xce = xc[(size_t)l * ED + e];
    float y = 0.f;
#pragma unroll
    for (int n = 0; n < NST; ++n) {
      float dA = __expf(dte * A[n]);
      float dBu = (dA - 1.f) / A[n] * sBm[s * NST + n] * xce;
      H[n] = dA * H[n] + dBu;
      y += H[n] * sCm[s * NST + n];
    }
    yout[(size_t)l * ED + e] = y;
  }
}

// ---------------------------------------------------------------- gate: (yf+yb+D*xc)*silu(z)
__global__ void k_gate(const float* __restrict__ yf, const float* __restrict__ yb,
                       const float* __restrict__ Dp, const float* __restrict__ xc,
                       const float* __restrict__ szg, bf16* __restrict__ ybf) {
  int idx = blockIdx.x * 256 + threadIdx.x;
  if (idx >= LSEQ * ED) return;
  int e = idx & (ED - 1);
  float y = yf[idx] + yb[idx] + Dp[e] * xc[idx];
  ybf[idx] = (bf16)(y * szg[idx]);
}

// ================================================================ host launch
extern "C" void kernel_launch(void* const* d_in, const int* in_sizes, int n_in,
                              void* d_out, int out_size, void* d_ws, size_t ws_size,
                              hipStream_t stream) {
  const float* kspace = (const float*)d_in[0];
  const float* mask   = (const float*)d_in[1];
  const float* pe_w   = (const float*)d_in[2];
  const float* pe_b   = (const float*)d_in[3];
  const float* pu_w   = (const float*)d_in[4];
  const float* pu_b   = (const float*)d_in[5];
  const float* ln_g   = (const float*)d_in[6];
  const float* ln_b   = (const float*)d_in[7];
  const float* fn_g   = (const float*)d_in[8];
  const float* fn_b   = (const float*)d_in[9];
  const float* cond_band   = (const float*)d_in[10];
  const float* cond_rad_w  = (const float*)d_in[11];
  const float* cond_mask_w = (const float*)d_in[12];
  const float* Win  = (const float*)d_in[13];
  const float* bin  = (const float*)d_in[14];
  const float* convw = (const float*)d_in[15];
  const float* convb = (const float*)d_in[16];
  const float* Wx   = (const float*)d_in[17];
  const float* Wdt  = (const float*)d_in[18];
  const float* bdt  = (const float*)d_in[19];
  const float* Alog = (const float*)d_in[20];
  const float* Dp   = (const float*)d_in[21];
  const float* Wout = (const float*)d_in[22];
  const float* bout = (const float*)d_in[23];
  const float* edt  = (const float*)d_in[24];
  const float* eB   = (const float*)d_in[25];
  const float* eC   = (const float*)d_in[26];
  const float* wr   = (const float*)d_in[27];
  const float* wm   = (const float*)d_in[28];
  float* outp = (float*)d_out;

  size_t off = 0;
  auto alloc = [&](size_t bytes) -> char* {
    size_t o = (off + 255) & ~(size_t)255;
    off = o + bytes;
    return (char*)d_ws + o;
  };
  const size_t L = LSEQ;
  float* m_arr = (float*)alloc(L * 4);
  float* sdt   = (float*)alloc(L * 4);
  float* sB    = (float*)alloc(L * 4);
  float* sC    = (float*)alloc(L * 4);
  float* xraw  = (float*)alloc(L * DM * 4);
  float* h     = (float*)alloc(L * DM * 4);
  float* xp    = (float*)alloc(L * ED * 4);   // reused as yf after conv
  float* szg   = (float*)alloc(L * ED * 4);
  float* xc    = (float*)alloc(L * ED * 4);
  float* dbl   = (float*)alloc(L * 48 * 4);
  float* dt    = (float*)alloc(L * ED * 4);
  float* Bm    = (float*)alloc(L * NST * 4);
  float* Cm    = (float*)alloc(L * NST * 4);
  float* Pc    = (float*)alloc((size_t)2 * NCH * ED * NST * 4);
  float* Sc    = (float*)alloc((size_t)2 * NCH * ED * NST * 4);
  float* Hc    = (float*)alloc((size_t)2 * NCH * ED * NST * 4);
  float* yb    = (float*)alloc(L * ED * 4);
  float* h2    = (float*)alloc(L * DM * 4);
  bf16* Xin   = (bf16*)alloc(L * DM * 2);
  bf16* hln   = (bf16*)alloc(L * DM * 2);
  bf16* xcbf  = (bf16*)alloc(L * ED * 2);
  bf16* ybf   = (bf16*)alloc(L * ED * 2);
  bf16* h2bf  = (bf16*)alloc(L * DM * 2);
  bf16* W0t   = (bf16*)alloc(128 * 128 * 2);
  bf16* Wint  = (bf16*)alloc(512 * 128 * 2);
  bf16* Wxt   = (bf16*)alloc(48 * 256 * 2);
  bf16* Woutt = (bf16*)alloc(128 * 256 * 2);
  bf16* Wut   = (bf16*)alloc(128 * 128 * 2);
  float* yf = xp;   // conv consumer done before scan pass 3 writes

  // scalars, gather, weight casts
  k_scalars<<<(L + 255) / 256, 256, 0, stream>>>(mask, edt, eB, eC, wr, wm, m_arr, sdt, sB, sC);
  k_gather<<<(L * 128) / 256, 256, 0, stream>>>(kspace, Xin);
  k_cast_direct<<<64, 256, 0, stream>>>(pe_w, W0t, 128 * 128);
  k_castT<<<256, 256, 0, stream>>>(Win, Wint, 128, 512, 512);
  k_castT<<<48, 256, 0, stream>>>(Wx, Wxt, 256, 40, 48);
  k_castT<<<128, 256, 0, stream>>>(Wout, Woutt, 256, 128, 128);
  k_castT<<<64, 256, 0, stream>>>(pu_w, Wut, 128, 128, 128);

  // tiles per GEMM = (M/16) * ceil(N/64); 8 waves per block
  // G0: patch embed, then LN + cond -> h, hln         (512*2 tiles)
  gemm_wmma<4, 0><<<128, 256, 0, stream>>>(Xin, W0t, xraw, L, 128, 128,
                                           nullptr, nullptr, nullptr, nullptr, nullptr);
  k_ln<<<L / 8, 256, 0, stream>>>(xraw, pe_b, cond_band, cond_rad_w, cond_mask_w, m_arr,
                                  ln_g, ln_b, fn_g, fn_b, h, hln);
  // G1: in-proj + split/silu                          (512*8 tiles)
  gemm_wmma<4, 1><<<512, 256, 0, stream>>>(hln, Wint, xp, L, 512, 128,
                                           bin, nullptr, szg, nullptr, nullptr);
  // conv
  k_conv<<<(L * ED) / 256, 256, 0, stream>>>(xp, convw, convb, xc, xcbf);
  // G2: x-proj (N padded 40->48, 3 sub-tiles)         (512*1 tiles)
  gemm_wmma<3, 0><<<64, 256, 0, stream>>>(xcbf, Wxt, dbl, L, 48, 256,
                                          nullptr, nullptr, nullptr, nullptr, nullptr);
  // dt / Bm / Cm
  k_dtbc<<<(L * ED) / 256, 256, 0, stream>>>(dbl, Wdt, bdt, sdt, sB, sC, dt, Bm, Cm);
  // chunked bidirectional scan
  k_scan1<<<dim3(NCH, 2), 256, 0, stream>>>(dt, xc, Bm, Alog, Pc, Sc);
  k_scan2<<<dim3(1, 2), 256, 0, stream>>>(Pc, Sc, Hc);
  k_scan3<<<dim3(NCH, 2), 256, 0, stream>>>(dt, xc, Bm, Cm, Alog, Hc, yf, yb);
  // gate
  k_gate<<<(L * ED) / 256, 256, 0, stream>>>(yf, yb, Dp, xc, szg, ybf);
  // G4: out-proj + residual -> h2 (f32 + bf16)        (512*2 tiles)
  gemm_wmma<4, 2><<<128, 256, 0, stream>>>(ybf, Woutt, h2, L, 128, 256,
                                           bout, h, nullptr, h2bf, nullptr);
  // G5: unembed + scatter into k-space residual       (512*2 tiles)
  gemm_wmma<4, 3><<<128, 256, 0, stream>>>(h2bf, Wut, outp, L, 128, 128,
                                           nullptr, kspace, nullptr, nullptr, pu_b);
  (void)in_sizes; (void)n_in; (void)out_size; (void)ws_size;
}
